// CGCNN_14078902797062
// MI455X (gfx1250) — matrix operand order repaired and checked by reading, not compiled, not run on hardware
//
#include <hip/hip_runtime.h>
#include <hip/hip_bf16.h>

// ---------------- types ----------------
typedef __attribute__((ext_vector_type(16))) __bf16 v16bf;
typedef __attribute__((ext_vector_type(8)))  float  v8f;

struct U4x2 { uint4 a, b; };

__device__ __forceinline__ v8f vzero8() {
    v8f z = {0.f,0.f,0.f,0.f,0.f,0.f,0.f,0.f};
    return z;
}

__device__ __forceinline__ unsigned short f2bf(float f) {
    unsigned int u = __float_as_uint(f);
    unsigned int r = u + 0x7FFFu + ((u >> 16) & 1u);   // RNE
    return (unsigned short)(r >> 16);
}

__device__ __forceinline__ float sigf(float x)  { return 1.0f / (1.0f + __expf(-x)); }
__device__ __forceinline__ float softp(float x) { return (x > 20.0f) ? x : __logf(1.0f + __expf(x)); }

// load a 32-byte (16 x bf16) fragment from two 16B chunks `gap` ushorts apart
__device__ __forceinline__ v16bf load_frag2(const unsigned short* p, int gap) {
    U4x2 t;
    t.a = *(const uint4*)(p);
    t.b = *(const uint4*)(p + gap);
    return __builtin_bit_cast(v16bf, t);
}

// CDNA5 async copy: global -> LDS, 16B per lane, tracked by ASYNCcnt
__device__ __forceinline__ void async_load_b128(unsigned lds_off, const void* gaddr) {
    asm volatile("global_load_async_to_lds_b128 %0, %1, off"
                 :: "v"(lds_off), "v"(gaddr) : "memory");
}
__device__ __forceinline__ void wait_asynccnt0() {
    asm volatile("s_wait_asynccnt 0x0" ::: "memory");
}
__device__ __forceinline__ void wait_dscnt0() {
    asm volatile("s_wait_dscnt 0x0" ::: "memory");
}

// ---------------- constants ----------------
#define DD   64
#define ZPAD 192          // 178 padded to 6 K-steps of 32
#define KPRE 96           // 92 padded to 3 K-steps of 32
#define WPB  8            // waves per block
#define ZROW 200          // bf16 elems per z row in LDS (400B, conflict-padded)
#define XROW 104          // bf16 elems per x row in LDS (208B)

// ---------------- utility kernels ----------------
__global__ void zero_f32_kernel(float* __restrict__ p, size_t n) {
    size_t i = (size_t)blockIdx.x * blockDim.x + threadIdx.x;
    size_t st = (size_t)gridDim.x * blockDim.x;
    for (; i < n; i += st) p[i] = 0.0f;
}

// Convert weights to WMMA-B fragment-native bf16 layout:
//   index = ((s*2 + hb)*64 + n)*16 + j  maps to W[K = s*32 + hb*16 + j][n]
__global__ void wprep_kernel(const float* __restrict__ Wpre,
                             const float* __restrict__ Wf,
                             const float* __restrict__ Ws,
                             unsigned short* __restrict__ WpF,
                             unsigned short* __restrict__ WfF,
                             unsigned short* __restrict__ WsF) {
    int tid = blockIdx.x * blockDim.x + threadIdx.x;
    int stride = gridDim.x * blockDim.x;
    for (int i = tid; i < KPRE * DD; i += stride) {
        int j = i & 15, n = (i >> 4) & 63, sh = i >> 10;
        int K = (sh >> 1) * 32 + (sh & 1) * 16 + j;
        float v = (K < 92) ? Wpre[(size_t)K * DD + n] : 0.0f;
        WpF[i] = f2bf(v);
    }
    for (int l = 0; l < 3; ++l) {
        for (int i = tid; i < ZPAD * DD; i += stride) {
            int j = i & 15, n = (i >> 4) & 63, sh = i >> 10;
            int K = (sh >> 1) * 32 + (sh & 1) * 16 + j;
            float vf = (K < 178) ? Wf[((size_t)l * 178 + K) * DD + n] : 0.0f;
            float vs = (K < 178) ? Ws[((size_t)l * 178 + K) * DD + n] : 0.0f;
            WfF[(size_t)l * (ZPAD * DD) + i] = f2bf(vf);
            WsF[(size_t)l * (ZPAD * DD) + i] = f2bf(vs);
        }
    }
}

// ea = edge_attr * gate  (bf16, padded 50 -> 64 cols), plus deg accumulation
__global__ void gate_kernel(const float* __restrict__ edge_attr,
                            const float* __restrict__ edge_dist,
                            const float* __restrict__ rmin_raw,
                            const float* __restrict__ rdelta_raw,
                            const int*   __restrict__ edge_index,
                            unsigned short* __restrict__ eabf,
                            float* __restrict__ deg, int E) {
    int gid = blockIdx.x * blockDim.x + threadIdx.x;
    int e = gid >> 6, k = gid & 63;
    if (e >= E) return;
    float rmin = softp(rmin_raw[0]);
    float rmax = rmin + softp(rdelta_raw[0]);
    float d = edge_dist[e];
    float gate = sigf(10.0f * (d - rmin)) * sigf(10.0f * (rmax - d));
    float v = (k < 50) ? edge_attr[(size_t)e * 50 + k] * gate : 0.0f;
    eabf[(size_t)e * 64 + k] = f2bf(v);
    if (k == 0) atomicAdd(&deg[edge_index[E + e]], 1.0f);
}

// h = relu(x @ W_pre + b_pre), WMMA bf16, writes f32 h and bf16 mirror
__global__ __launch_bounds__(256) void pre_kernel(const float* __restrict__ x,
                                                  const unsigned short* __restrict__ WpF,
                                                  const float* __restrict__ bpre,
                                                  float* __restrict__ h,
                                                  unsigned short* __restrict__ hbf, int N) {
    __shared__ __align__(16) unsigned short xsm[WPB * 16 * XROW];
    const int lane = threadIdx.x & 31;
    const int wave = threadIdx.x >> 5;
    const int nodeBase = (blockIdx.x * WPB + wave) * 16;
    unsigned short* xw = xsm + wave * (16 * XROW);

    // wave-private staging with f32->bf16 conversion
    for (int i = lane; i < 16 * KPRE; i += 32) {
        int row = i / KPRE, k = i - row * KPRE;
        int node = nodeBase + row;
        float v = (node < N && k < 92) ? x[(size_t)node * 92 + k] : 0.0f;
        xw[row * XROW + k] = f2bf(v);
    }
    wait_dscnt0();   // tile is wave-private: wave-local LDS completion is enough

    const int hb = lane >> 4, col = lane & 15, rowA = lane & 15;
    v8f acc[4];
#pragma unroll
    for (int t = 0; t < 4; ++t) acc[t] = vzero8();

#pragma unroll
    for (int s = 0; s < 3; ++s) {
        const unsigned short* ap = xw + rowA * XROW + s * 32 + hb * 16;
        v16bf a = load_frag2(ap, 8);
#pragma unroll
        for (int t = 0; t < 4; ++t) {
            const unsigned short* bp = WpF + ((size_t)((s * 2 + hb) * 64 + t * 16 + col)) * 16;
            v16bf b = load_frag2(bp, 8);
            acc[t] = __builtin_amdgcn_wmma_f32_16x16x32_bf16(false, a, false, b,
                                                             (short)0, acc[t], false, false);
        }
    }
#pragma unroll
    for (int t = 0; t < 4; ++t) {
        int n = t * 16 + col;
        float bb = bpre[n];
#pragma unroll
        for (int r = 0; r < 8; ++r) {
            int node = nodeBase + r + 8 * hb;
            if (node < N) {
                float v = fmaxf(acc[t][r] + bb, 0.0f);
                h[(size_t)node * DD + n] = v;
                hbf[(size_t)node * DD + n] = f2bf(v);
            }
        }
    }
}

// message GEMM: z=(h[dst],h[src],ea) @ {Wf,Ws}, sigmoid*softplus, scatter into msum
__global__ __launch_bounds__(256) void msg_kernel(const unsigned short* __restrict__ hbf,
                                                  const unsigned short* __restrict__ eabf,
                                                  const int* __restrict__ edge_index,
                                                  const unsigned short* __restrict__ WfF,
                                                  const unsigned short* __restrict__ WsF,
                                                  const float* __restrict__ bfv,
                                                  const float* __restrict__ bsv,
                                                  float* __restrict__ msum, int E) {
    __shared__ __align__(16) unsigned short zsm[WPB * 16 * ZROW];
    const int lane = threadIdx.x & 31;
    const int wave = threadIdx.x >> 5;
    const int tileBase = (blockIdx.x * WPB + wave) * 16;
    unsigned short* zw = zsm + wave * (16 * ZROW);
    const unsigned ldsBase = (unsigned)(uintptr_t)zw;  // LDS aperture: low 32 bits = offset

    // stage 16 z rows (3 segments x 128B each) via async global->LDS b128,
    // 16B per lane per issue, 12 issues per lane, tracked by ASYNCcnt
    for (int c = lane; c < 16 * 3 * 8; c += 32) {
        int chunk = c & 7;
        int seg = c >> 3;                 // 0..47
        int row = seg / 3, part = seg - row * 3;
        int e = tileBase + row;
        const unsigned short* src;
        if (part == 0)      src = hbf + (size_t)edge_index[E + e] * 64;
        else if (part == 1) src = hbf + (size_t)edge_index[e] * 64;
        else                src = eabf + (size_t)e * 64;
        unsigned loff = ldsBase + (unsigned)(row * ZROW + part * 64 + chunk * 8) * 2u;
        async_load_b128(loff, src + chunk * 8);
    }
    wait_asynccnt0();   // tile is wave-private: no block barrier needed

    const int hb = lane >> 4, col = lane & 15, rowA = lane & 15;
    v8f accF[4], accS[4];
#pragma unroll
    for (int t = 0; t < 4; ++t) { accF[t] = vzero8(); accS[t] = vzero8(); }

#pragma unroll
    for (int s = 0; s < 6; ++s) {
        const unsigned short* ap = zw + rowA * ZROW + s * 32 + hb * 16;
        v16bf a = load_frag2(ap, 8);
#pragma unroll
        for (int t = 0; t < 4; ++t) {
            size_t boff = ((size_t)((s * 2 + hb) * 64 + t * 16 + col)) * 16;
            v16bf bF = load_frag2(WfF + boff, 8);
            accF[t] = __builtin_amdgcn_wmma_f32_16x16x32_bf16(false, a, false, bF,
                                                              (short)0, accF[t], false, false);
            v16bf bS = load_frag2(WsF + boff, 8);
            accS[t] = __builtin_amdgcn_wmma_f32_16x16x32_bf16(false, a, false, bS,
                                                              (short)0, accS[t], false, false);
        }
    }

#pragma unroll
    for (int t = 0; t < 4; ++t) {
        int n = t * 16 + col;
        float bfb = bfv[n], bsb = bsv[n];
#pragma unroll
        for (int r = 0; r < 8; ++r) {
            int e = tileBase + r + 8 * hb;
            float val = sigf(accF[t][r] + bfb) * softp(accS[t][r] + bsb);
            int d = edge_index[E + e];
            atomicAdd(&msum[(size_t)d * DD + n], val);
        }
    }
}

// h += msum/deg ; accumulate per-feature sum & sumsq into stats[0:64],[64:128]
__global__ void update_kernel(float* __restrict__ h, const float* __restrict__ msum,
                              const float* __restrict__ deg, float* __restrict__ stats, int N) {
    __shared__ float ssum[DD], ssq[DD];
    if (threadIdx.x < DD) { ssum[threadIdx.x] = 0.0f; ssq[threadIdx.x] = 0.0f; }
    __syncthreads();
    size_t total = (size_t)N * DD;
    size_t st = (size_t)gridDim.x * blockDim.x;
    for (size_t i = (size_t)blockIdx.x * blockDim.x + threadIdx.x; i < total; i += st) {
        int node = (int)(i >> 6), f = (int)(i & 63);
        float dg = deg[node]; dg = dg < 1.0f ? 1.0f : dg;
        float v = h[i] + msum[i] / dg;
        h[i] = v;
        atomicAdd(&ssum[f], v);
        atomicAdd(&ssq[f], v * v);
    }
    __syncthreads();
    if (threadIdx.x < DD) {
        atomicAdd(&stats[threadIdx.x], ssum[threadIdx.x]);
        atomicAdd(&stats[DD + threadIdx.x], ssq[threadIdx.x]);
    }
}

// layernorm-over-nodes; refresh bf16 mirror
__global__ void norm_kernel(float* __restrict__ h, unsigned short* __restrict__ hbf,
                            const float* __restrict__ stats,
                            const float* __restrict__ gam, const float* __restrict__ bet, int N) {
    size_t total = (size_t)N * DD;
    size_t st = (size_t)gridDim.x * blockDim.x;
    float invN = 1.0f / (float)N;
    for (size_t i = (size_t)blockIdx.x * blockDim.x + threadIdx.x; i < total; i += st) {
        int f = (int)(i & 63);
        float mu = stats[f] * invN;
        float var = stats[DD + f] * invN - mu * mu;
        float inv = rsqrtf(var + 1e-5f);
        float v = gam[f] * (h[i] - mu) * inv + bet[f];
        h[i] = v;
        hbf[i] = f2bf(v);
    }
}

__global__ void pool_kernel(const float* __restrict__ h, const int* __restrict__ batch,
                            float* __restrict__ pooled, float* __restrict__ cnt, int N) {
    size_t total = (size_t)N * DD;
    size_t st = (size_t)gridDim.x * blockDim.x;
    for (size_t i = (size_t)blockIdx.x * blockDim.x + threadIdx.x; i < total; i += st) {
        int node = (int)(i >> 6), f = (int)(i & 63);
        int g = batch[node];
        atomicAdd(&pooled[(size_t)g * DD + f], h[i]);
        if (f == 0) atomicAdd(&cnt[g], 1.0f);
    }
}

// emb = relu(pooled_mean @ W_post + b_post); out = emb @ W_out + b_out
// d_out layout: [out(512) | emb(512*64)]
__global__ __launch_bounds__(64) void head_kernel(const float* __restrict__ pooled,
                                                  const float* __restrict__ cnt,
                                                  const float* __restrict__ Wpost,
                                                  const float* __restrict__ bpost,
                                                  const float* __restrict__ Wout,
                                                  const float* __restrict__ bout,
                                                  float* __restrict__ out, int G) {
    __shared__ float pm[DD];
    __shared__ float er[DD];
    int g = blockIdx.x, j = threadIdx.x;
    float c = cnt[g]; c = c < 1.0f ? 1.0f : c;
    pm[j] = pooled[(size_t)g * DD + j] / c;
    __syncthreads();
    float acc = bpost[j];
#pragma unroll 8
    for (int k = 0; k < DD; ++k) acc += pm[k] * Wpost[(size_t)k * DD + j];
    float e = fmaxf(acc, 0.0f);
    out[(size_t)G + (size_t)g * DD + j] = e;
    er[j] = e * Wout[j];
    __syncthreads();
    for (int off = 32; off > 0; off >>= 1) {
        if (j < off) er[j] += er[j + off];
        __syncthreads();
    }
    if (j == 0) out[g] = er[0] + bout[0];
}

// ---------------- orchestration ----------------
extern "C" void kernel_launch(void* const* d_in, const int* in_sizes, int n_in,
                              void* d_out, int out_size, void* d_ws, size_t ws_size,
                              hipStream_t stream) {
    const float* x          = (const float*)d_in[0];
    const int*   edge_index = (const int*)d_in[1];
    const float* edge_attr  = (const float*)d_in[2];
    const float* edge_dist  = (const float*)d_in[3];
    const int*   batch      = (const int*)d_in[4];
    const float* W_pre      = (const float*)d_in[5];
    const float* b_pre      = (const float*)d_in[6];
    const float* Wf         = (const float*)d_in[7];
    const float* bfp        = (const float*)d_in[8];
    const float* Ws         = (const float*)d_in[9];
    const float* bsp        = (const float*)d_in[10];
    const float* gamma      = (const float*)d_in[11];
    const float* beta       = (const float*)d_in[12];
    const float* rmin_raw   = (const float*)d_in[13];
    const float* rdelta_raw = (const float*)d_in[14];
    const float* W_post     = (const float*)d_in[15];
    const float* b_post     = (const float*)d_in[16];
    const float* W_out      = (const float*)d_in[17];
    const float* b_out      = (const float*)d_in[18];

    const int E = in_sizes[3];
    const int N = in_sizes[4];
    const int G = 512;

    size_t off = 0;
    auto carve = [&](size_t bytes) { size_t o = off; off += (bytes + 255) & ~(size_t)255; return o; };
    char* ws = (char*)d_ws;
    float*          h     = (float*)(ws + carve((size_t)N * DD * 4));
    unsigned short* hbf   = (unsigned short*)(ws + carve((size_t)N * DD * 2));
    unsigned short* eabf  = (unsigned short*)(ws + carve((size_t)E * 64 * 2));
    float*          msum  = (float*)(ws + carve((size_t)N * DD * 4));
    float*          deg   = (float*)(ws + carve((size_t)N * 4));
    unsigned short* WpF   = (unsigned short*)(ws + carve((size_t)KPRE * DD * 2));
    unsigned short* WfF   = (unsigned short*)(ws + carve((size_t)3 * ZPAD * DD * 2));
    unsigned short* WsF   = (unsigned short*)(ws + carve((size_t)3 * ZPAD * DD * 2));
    float*          stats = (float*)(ws + carve(2 * DD * 4));
    float*          pooled= (float*)(ws + carve((size_t)G * DD * 4));
    float*          cnt   = (float*)(ws + carve((size_t)G * 4));

    const int T = 256;
    zero_f32_kernel<<<(N + T - 1) / T, T, 0, stream>>>(deg, (size_t)N);
    zero_f32_kernel<<<(G * DD + T - 1) / T, T, 0, stream>>>(pooled, (size_t)G * DD);
    zero_f32_kernel<<<(G + T - 1) / T, T, 0, stream>>>(cnt, (size_t)G);

    wprep_kernel<<<64, T, 0, stream>>>(W_pre, Wf, Ws, WpF, WfF, WsF);
    {
        long long total = (long long)E * 64;
        gate_kernel<<<(int)((total + T - 1) / T), T, 0, stream>>>(
            edge_attr, edge_dist, rmin_raw, rdelta_raw, edge_index, eabf, deg, E);
    }

    pre_kernel<<<(N + 16 * WPB - 1) / (16 * WPB), T, 0, stream>>>(x, WpF, b_pre, h, hbf, N);

    const int elemBlocks = (int)(((size_t)N * DD + T - 1) / T);
    for (int l = 0; l < 3; ++l) {
        zero_f32_kernel<<<elemBlocks, T, 0, stream>>>(msum, (size_t)N * DD);
        zero_f32_kernel<<<1, 128, 0, stream>>>(stats, (size_t)(2 * DD));
        msg_kernel<<<E / (16 * WPB), T, 0, stream>>>(
            hbf, eabf, edge_index,
            WfF + (size_t)l * ZPAD * DD, WsF + (size_t)l * ZPAD * DD,
            bfp + (size_t)l * DD, bsp + (size_t)l * DD, msum, E);
        update_kernel<<<1024, T, 0, stream>>>(h, msum, deg, stats, N);
        norm_kernel<<<elemBlocks, T, 0, stream>>>(h, hbf, stats,
                                                  gamma + (size_t)l * DD, beta + (size_t)l * DD, N);
    }

    pool_kernel<<<elemBlocks, T, 0, stream>>>(h, batch, pooled, cnt, N);
    head_kernel<<<G, 64, 0, stream>>>(pooled, cnt, W_post, b_post, W_out, b_out,
                                      (float*)d_out, G);
}
